// Net_32461362823757
// MI455X (gfx1250) — matrix operand order repaired and checked by reading, not compiled, not run on hardware
//
#include <hip/hip_runtime.h>
#include <stdint.h>

#define B_   128
#define T_   1024
#define DIN  128
#define DH   256
#define DL   512
#define G4   1024
#define KTOT 768
#define TS_  8
#define DOUT 8

typedef __attribute__((ext_vector_type(16))) __bf16 v16bf;
typedef __attribute__((ext_vector_type(8)))  float  v8f;

struct alignas(16) U4 { unsigned int x, y, z, w; };
union Frag { U4 u[2]; v16bf v; };

__device__ __forceinline__ unsigned short f2bf(float f) {
  unsigned int u = __builtin_bit_cast(unsigned int, f);
  u += 0x7FFFu + ((u >> 16) & 1u);
  return (unsigned short)(u >> 16);
}
__device__ __forceinline__ float bf2f(unsigned short b) {
  return __builtin_bit_cast(float, (unsigned int)b << 16);
}
__device__ __forceinline__ float fsigmoid(float x) {
  return __builtin_amdgcn_rcpf(1.f + __expf(-x));
}
__device__ __forceinline__ float ftanh(float x) {
  return 2.f * __builtin_amdgcn_rcpf(1.f + __expf(-2.f * x)) - 1.f;
}

// ---------------- weight prep: f32 -> bf16, fuse [Wih|Whh], bih+bhh ----------------
__global__ void prep_kernel(const float* __restrict__ W_in,
                            const float* __restrict__ Wih1, const float* __restrict__ Whh1,
                            const float* __restrict__ bih1, const float* __restrict__ bhh1,
                            const float* __restrict__ Wih2, const float* __restrict__ Whh2,
                            const float* __restrict__ bih2, const float* __restrict__ bhh2,
                            unsigned short* __restrict__ Wb_in,
                            unsigned short* __restrict__ wcomb,
                            float* __restrict__ bsum) {
  const int i = blockIdx.x * blockDim.x + threadIdx.x;
  if (i < DL * DIN) Wb_in[i] = f2bf(W_in[i]);
  const int NW = G4 * KTOT;
  if (i < 2 * NW) {
    const int d = i / NW;
    const int r = i - d * NW;
    const int n = r / KTOT;
    const int k = r - n * KTOT;
    const float* Wih = d ? Wih2 : Wih1;
    const float* Whh = d ? Whh2 : Whh1;
    const float v = (k < DL) ? Wih[n * DL + k] : Whh[n * DH + (k - DL)];
    wcomb[i] = f2bf(v);
  }
  if (i < 2 * G4) {
    const int d = i >> 10, n = i & (G4 - 1);
    bsum[i] = d ? (bih2[n] + bhh2[n]) : (bih1[n] + bhh1[n]);
  }
}

// ---------------- input GEMM: seq0[t][b][n] = x[b][t][:] @ W_in.T + b_in (bf16 WMMA) -------
__global__ __launch_bounds__(256, 1) void in_gemm_kernel(
    const float* __restrict__ x, const unsigned short* __restrict__ Wb,
    const float* __restrict__ b_in, unsigned short* __restrict__ seq0) {
  const int t = blockIdx.x;
  const int nb = blockIdx.y;  // 128-col chunk of the 512 outputs
  __shared__ alignas(16) unsigned short xs[B_ * DIN];  // 32 KB bf16 staging
  const int tid = threadIdx.x;
  for (int i = tid; i < B_ * DIN; i += 256) {
    const int b = i >> 7, k = i & 127;
    xs[i] = f2bf(x[((size_t)b * T_ + t) * DIN + k]);
  }
  __syncthreads();
  const int wave = tid >> 5, lane = tid & 31, lhalf = lane >> 4, lmod = lane & 15;
  const int nt = nb * 8 + wave;  // N-tile 0..31
  const v8f zero8 = {0.f, 0.f, 0.f, 0.f, 0.f, 0.f, 0.f, 0.f};
  v8f acc[8];
#pragma unroll
  for (int mt = 0; mt < 8; ++mt) acc[mt] = zero8;
  const size_t brow = (size_t)(nt * 16 + lmod) * DIN + lhalf * 16;
#pragma unroll
  for (int kt = 0; kt < 4; ++kt) {
    Frag Bf;
    const U4* p = (const U4*)(Wb + brow + kt * 32);
    Bf.u[0] = p[0];
    Bf.u[1] = p[1];
#pragma unroll
    for (int mt = 0; mt < 8; ++mt) {
      Frag Af;
      const int m = mt * 16 + lmod;
      const int k0 = kt * 32 + (lhalf ? 16 : 0);
      Af.u[0] = *(const U4*)(&xs[m * DIN + k0]);
      Af.u[1] = *(const U4*)(&xs[m * DIN + k0 + 8]);
      acc[mt] = __builtin_amdgcn_wmma_f32_16x16x32_bf16(
          false, Af.v, false, Bf.v, (short)0, acc[mt], false, false);
    }
  }
  const int n = nt * 16 + lmod;
  const float bias = b_in[n];
#pragma unroll
  for (int mt = 0; mt < 8; ++mt)
#pragma unroll
    for (int e = 0; e < 8; ++e) {
      const int m = mt * 16 + e + 8 * lhalf;
      seq0[((size_t)t * B_ + m) * DL + n] = f2bf(acc[mt][e] + bias);
    }
}

// ---------------- persistent LSTM layer: 1 workgroup per direction ----------------
// gates = [x_t | h] @ [Wih;Whh].T + (bih+bhh); fused WMMA over K=768; h lives in LDS.
__global__ __launch_bounds__(1024, 1) void lstm_layer_kernel(
    const unsigned short* __restrict__ seqIn,   // [T][128][512] bf16
    unsigned short* __restrict__ seqOut,        // [T][128][512] bf16
    const unsigned short* __restrict__ wcomb,   // [2][1024][768] bf16
    const float* __restrict__ bsum,             // [2][1024]
    const float* __restrict__ wt1, const float* __restrict__ wt2) {
  const int dir = blockIdx.x;  // 0 = forward, 1 = backward
  const unsigned short* W = wcomb + (size_t)dir * G4 * KTOT;
  const float* bs = bsum + dir * G4;
  const float* wt = dir ? wt2 : wt1;

  __shared__ alignas(16) unsigned short h_lds[B_ * DH];  // 64 KB

  const int tid = threadIdx.x;
  const int wave = tid >> 5;
  const int lane = tid & 31;
  const int jt = wave & 15;   // hidden tile (16 cols): owns gates i/f/g/o for this slice
  const int mg = wave >> 4;   // batch-row group: M-tiles mg*4 .. mg*4+3
  const int lhalf = lane >> 4;
  const int lmod = lane & 15;

  for (int i = tid; i < B_ * DH; i += 1024) h_lds[i] = 0;

  float bias[4];
  size_t brow[4];
#pragma unroll
  for (int g = 0; g < 4; ++g) {
    const int n = g * DH + jt * 16 + lmod;
    bias[g] = bs[n];
    brow[g] = (size_t)n * KTOT + lhalf * 16;
  }

  float wtv[TS_];
#pragma unroll
  for (int i = 0; i < TS_; ++i) wtv[i] = wt[i];

  const v8f zero8 = {0.f, 0.f, 0.f, 0.f, 0.f, 0.f, 0.f, 0.f};
  v8f c[4], hm[4], cm[4];
#pragma unroll
  for (int mt = 0; mt < 4; ++mt) { c[mt] = zero8; hm[mt] = zero8; cm[mt] = zero8; }

  __syncthreads();

  for (int t = 0; t < T_; ++t) {
    const int tin = dir ? (T_ - 1 - t) : t;
    const unsigned short* xrow = seqIn + (size_t)tin * B_ * DL;

    v8f acc[4][4];
#pragma unroll
    for (int g = 0; g < 4; ++g)
#pragma unroll
      for (int mt = 0; mt < 4; ++mt) acc[g][mt] = zero8;

#pragma unroll
    for (int kt = 0; kt < 24; ++kt) {  // K tiles: 0..15 from x (global), 16..23 from h (LDS)
      Frag Bf[4];
#pragma unroll
      for (int g = 0; g < 4; ++g) {
        const U4* p = (const U4*)(W + brow[g] + kt * 32);
        Bf[g].u[0] = p[0];
        Bf[g].u[1] = p[1];
      }
      Frag Af[4];
      const int koff = kt * 32 + (lhalf ? 16 : 0);
#pragma unroll
      for (int mt = 0; mt < 4; ++mt) {
        const int m = (mg * 4 + mt) * 16 + lmod;
        if (kt < 16) {
          Af[mt].u[0] = *(const U4*)(xrow + (size_t)m * DL + koff);
          Af[mt].u[1] = *(const U4*)(xrow + (size_t)m * DL + koff + 8);
        } else {
          const int kh = (kt - 16) * 32 + (lhalf ? 16 : 0);
          Af[mt].u[0] = *(const U4*)(&h_lds[m * DH + kh]);
          Af[mt].u[1] = *(const U4*)(&h_lds[m * DH + kh + 8]);
        }
      }
#pragma unroll
      for (int g = 0; g < 4; ++g)
#pragma unroll
        for (int mt = 0; mt < 4; ++mt)
          acc[g][mt] = __builtin_amdgcn_wmma_f32_16x16x32_bf16(
              false, Af[mt].v, false, Bf[g].v, (short)0, acc[g][mt], false, false);
    }

    __syncthreads();  // all reads of h_{t-1} complete before overwrite

    const int tc = t & (TS_ - 1);
    const float w8 = wtv[tc];
    const bool merge = (tc == TS_ - 1);
    const int j = jt * 16 + lmod;

#pragma unroll
    for (int mt = 0; mt < 4; ++mt) {
#pragma unroll
      for (int e = 0; e < 8; ++e) {
        const float gi = fsigmoid(acc[0][mt][e] + bias[0]);
        const float gf = fsigmoid(acc[1][mt][e] + bias[1]);
        const float gg = ftanh(acc[2][mt][e] + bias[2]);
        const float go = fsigmoid(acc[3][mt][e] + bias[3]);
        const float cn = gf * c[mt][e] + gi * gg;
        const float hn = go * ftanh(cn);
        const float hmv = hm[mt][e] + w8 * hn;
        const float cmv = cm[mt][e] + w8 * cn;
        const float hsel = merge ? hmv : hn;   // merged h replaces carry AND output at chunk end
        c[mt][e]  = merge ? cmv : cn;
        hm[mt][e] = merge ? 0.f : hmv;
        cm[mt][e] = merge ? 0.f : cmv;
        const int m = (mg * 4 + mt) * 16 + e + 8 * lhalf;
        const unsigned short hb = f2bf(hsel);
        h_lds[m * DH + j] = hb;
        seqOut[((size_t)t * B_ + m) * DL + dir * DH + j] = hb;  // processing order, per ref
      }
    }
    __syncthreads();  // h_t visible before next step's reads

    if (t + 1 < T_) {  // hide HBM latency of the x stream (global_prefetch_b8)
      const int tin2 = dir ? (T_ - 2 - t) : (t + 1);
      __builtin_prefetch(seqIn + (size_t)tin2 * B_ * DL +
                             (size_t)((mg * 4) * 16 + lmod) * DL + lane,
                         0, 3);
    }
  }
}

// ---------------- output head: logits[b][d][t] = seq[t][b][:] @ W_out.T + b_out ----------
__global__ void logits_kernel(const unsigned short* __restrict__ seq,
                              const float* __restrict__ Wout,
                              const float* __restrict__ bout,
                              float* __restrict__ logit) {
  const int idx = blockIdx.x * 256 + threadIdx.x;
  if (idx >= B_ * T_ * DOUT) return;
  const int d = idx & 7;
  const int t = (idx >> 3) & (T_ - 1);
  const int b = idx >> 13;
  const unsigned short* row = seq + ((size_t)t * B_ + b) * DL;
  const float* w = Wout + d * DL;
  float s = bout[d];
#pragma unroll 8
  for (int k = 0; k < DL; ++k) s += bf2f(row[k]) * w[k];
  logit[((size_t)b * DOUT + d) * T_ + t] = s;  // time-contiguous for the softmax reduction
}

// ---------------- log_softmax over time (axis=1), one block per (b,d) ----------------
__global__ __launch_bounds__(256) void lsm_kernel(const float* __restrict__ logit,
                                                  float* __restrict__ out) {
  const int b = blockIdx.x >> 3;
  const int d = blockIdx.x & 7;
  const float* p = logit + ((size_t)b * DOUT + d) * T_;
  __shared__ float red[256];
  const int tid = threadIdx.x;
  float mx = -3.4e38f;
  for (int t = tid; t < T_; t += 256) mx = fmaxf(mx, p[t]);
  red[tid] = mx;
  __syncthreads();
  for (int s = 128; s > 0; s >>= 1) {
    if (tid < s) red[tid] = fmaxf(red[tid], red[tid + s]);
    __syncthreads();
  }
  const float m = red[0];
  __syncthreads();
  float sum = 0.f;
  for (int t = tid; t < T_; t += 256) sum += __expf(p[t] - m);
  red[tid] = sum;
  __syncthreads();
  for (int s = 128; s > 0; s >>= 1) {
    if (tid < s) red[tid] += red[tid + s];
    __syncthreads();
  }
  const float lse = m + __logf(red[0]);
  for (int t = tid; t < T_; t += 256)
    out[((size_t)b * T_ + t) * DOUT + d] = p[t] - lse;
}

extern "C" void kernel_launch(void* const* d_in, const int* in_sizes, int n_in,
                              void* d_out, int out_size, void* d_ws, size_t ws_size,
                              hipStream_t stream) {
  (void)in_sizes; (void)n_in; (void)out_size; (void)ws_size;
  const float* x     = (const float*)d_in[0];
  const float* W_in  = (const float*)d_in[1];
  const float* b_in  = (const float*)d_in[2];
  const float* Wih1  = (const float*)d_in[3];
  const float* Whh1  = (const float*)d_in[4];
  const float* bih1  = (const float*)d_in[5];
  const float* bhh1  = (const float*)d_in[6];
  const float* Wih2  = (const float*)d_in[7];
  const float* Whh2  = (const float*)d_in[8];
  const float* bih2  = (const float*)d_in[9];
  const float* bhh2  = (const float*)d_in[10];
  const float* wt1   = (const float*)d_in[11];
  const float* wt2   = (const float*)d_in[12];
  const float* W_out = (const float*)d_in[13];
  const float* b_out = (const float*)d_in[14];
  float* out = (float*)d_out;

  char* ws = (char*)d_ws;
  const size_t SEQ = (size_t)T_ * B_ * DL * sizeof(unsigned short);  // 128 MB
  unsigned short* seq0  = (unsigned short*)(ws);
  unsigned short* seq1  = (unsigned short*)(ws + SEQ);
  size_t off = 2 * SEQ;
  unsigned short* Wb_in = (unsigned short*)(ws + off); off += (size_t)DL * DIN * 2;
  unsigned short* wcomb = (unsigned short*)(ws + off); off += (size_t)2 * G4 * KTOT * 2;
  float* bsum = (float*)(ws + off); off += (size_t)2 * G4 * 4;
  float* logit = (float*)(ws + off);

  prep_kernel<<<(2 * G4 * KTOT + 255) / 256, 256, 0, stream>>>(
      W_in, Wih1, Whh1, bih1, bhh1, Wih2, Whh2, bih2, bhh2, Wb_in, wcomb, bsum);
  in_gemm_kernel<<<dim3(T_, 4), 256, 0, stream>>>(x, Wb_in, b_in, seq0);
  // Layer 1: seq0 -> seq1 ; Layer 2: seq1 -> seq0 (same weights, per reference)
  lstm_layer_kernel<<<2, 1024, 0, stream>>>(seq0, seq1, wcomb, bsum, wt1, wt2);
  lstm_layer_kernel<<<2, 1024, 0, stream>>>(seq1, seq0, wcomb, bsum, wt1, wt2);
  logits_kernel<<<(B_ * T_ * DOUT + 255) / 256, 256, 0, stream>>>(seq0, W_out, b_out, logit);
  lsm_kernel<<<B_ * DOUT, 256, 0, stream>>>(logit, out);
}
